// Kan1xN_61710090109687
// MI455X (gfx1250) — matrix operand order repaired and checked by reading, not compiled, not run on hardware
//
#include <hip/hip_runtime.h>
#include <hip/hip_bf16.h>
#include <math.h>

typedef __attribute__((ext_vector_type(16))) _Float16 v16h;
typedef __attribute__((ext_vector_type(8)))  float    v8f;
typedef __attribute__((ext_vector_type(4)))  float    v4f;

#define NPB   128   // points per block
#define NOUT  128
#define NPAR  64

__device__ __forceinline__ float knotv(int i) {
    // knots: [LB,LB,LB, LB + 0.1*(i-3) for i=3..63, UB,UB,UB]
    int c = i - 3;
    c = c < 0 ? 0 : (c > 60 ? 60 : c);
    return -3.0f + 0.1f * (float)c;
}

template <bool GUARD>
__global__ __launch_bounds__(256) void kan_wmma_kernel(
    const float* __restrict__ x, const float* __restrict__ params,
    float* __restrict__ out, int npts)
{
    // A staged in WMMA 16-bit fragment layout:
    // Afrag[t(8)][kb2(2)][lane(32)][16 halves]  -> 16 KB
    __shared__ __align__(32) _Float16 Afrag[8 * 2 * 32 * 16];

    const int tid   = threadIdx.x;
    const int lane  = tid & 31;
    const int wave  = tid >> 5;
    const int pbase = blockIdx.x * NPB;

    // ---- zero A staging (16 KB, coalesced b128 stores) ----
    {
        v4f z = {0.f, 0.f, 0.f, 0.f};
        v4f* zp = (v4f*)Afrag;
        #pragma unroll
        for (int i = 0; i < 4; ++i) zp[tid + 256 * i] = z;
    }
    __syncthreads();

    // ---- phase 1: one point per thread (threads 0..127), scatter to LDS ----
    if (tid < NPB) {
        int p  = pbase + tid;
        int pc = (GUARD && p >= npts) ? (npts - 1) : p;
        float xv = x[pc];
        const int t = tid >> 4, m = tid & 15;

        // silu(x) -> basis column 63
        float s = xv / (1.0f + __expf(-xv));

        // interval index: t[j] <= x < t[j+1], j in [3,62]
        bool valid = (xv >= -3.0f) && (xv < 3.0f);
        int j = 3 + (int)floorf((xv + 3.0f) * 10.0f);
        j = j < 3 ? 3 : (j > 62 ? 62 : j);

        // de Boor: 4 nonzero cubic basis values B_{j-3..j,3}(x)
        float Nv[4] = {1.0f, 0.0f, 0.0f, 0.0f};
        float lft[4], rgt[4];
        #pragma unroll
        for (int d = 1; d <= 3; ++d) {
            lft[d] = xv - knotv(j + 1 - d);
            rgt[d] = knotv(j + d) - xv;
            float saved = 0.0f;
            #pragma unroll
            for (int r = 0; r < d; ++r) {
                float tmp = Nv[r] / (rgt[r + 1] + lft[d - r]);
                Nv[r] = saved + rgt[r + 1] * tmp;
                saved = lft[d - r] * tmp;
            }
            Nv[d] = saved;
        }

        // scatter into 16-bit A-matrix fragment layout (ISA 7.12.2):
        // lane = m + 16*((k>>3)&1); half = ( ((k>>4)<<2) + ((k&7)>>1) )*2 + (k&1)
        auto scat = [&](int K, float val) {
            int kb2 = K >> 5, k = K & 31;
            int ln  = m + (((k >> 3) & 1) << 4);
            int v   = ((k >> 4) << 2) + ((k & 7) >> 1);
            int h   = k & 1;
            Afrag[(((t * 2 + kb2) * 32 + ln) << 4) + (v * 2 + h)] = (_Float16)val;
        };
        if (valid) {
            #pragma unroll
            for (int i = 0; i < 4; ++i) scat(j - 3 + i, Nv[i]);
        }
        scat(63, s);
    }

    // ---- B fragments: params columns for this wave, f32 -> f16 in regs ----
    // 16-bit B layout: lanes 0-15 hold K=0..7,16..23, lanes 16-31 hold K=8..15,24..31
    const int n0   = wave * 16;
    const int ncol = n0 + (lane & 15);
    const int khi  = (lane >> 4) * 8;
    v16h b0, b1;
    #pragma unroll
    for (int e = 0; e < 16; ++e) {
        b0[e] = (_Float16)params[(khi + (e & 7) + ((e >> 3) << 4)) * NOUT + ncol];
        b1[e] = (_Float16)params[(32 + khi + (e & 7) + ((e >> 3) << 4)) * NOUT + ncol];
    }
    __syncthreads();

    // ---- phase 2: WMMA over 8 point-tiles, 2 tiles in flight to hide the
    //      F16-WMMA RAW hazard window with independent WMMAs ----
    const v16h* Av = (const v16h*)Afrag;
    const int mrow = (lane >> 4) * 8;                 // C/D: hi half-wave -> M+8
    float* outrow = out + (size_t)ncol * (size_t)npts + pbase + mrow;

    #pragma unroll
    for (int t = 0; t < 8; t += 2) {
        v16h a00 = Av[(t * 2 + 0) * 32 + lane];
        v16h a01 = Av[(t * 2 + 1) * 32 + lane];
        v16h a10 = Av[(t * 2 + 2) * 32 + lane];
        v16h a11 = Av[(t * 2 + 3) * 32 + lane];
        v8f c0 = {}, c1 = {};
        c0 = __builtin_amdgcn_wmma_f32_16x16x32_f16(false, a00, false, b0,
                                                    (short)0, c0, false, false);
        c1 = __builtin_amdgcn_wmma_f32_16x16x32_f16(false, a10, false, b0,
                                                    (short)0, c1, false, false);
        c0 = __builtin_amdgcn_wmma_f32_16x16x32_f16(false, a01, false, b1,
                                                    (short)0, c0, false, false);
        c1 = __builtin_amdgcn_wmma_f32_16x16x32_f16(false, a11, false, b1,
                                                    (short)0, c1, false, false);

        float* dst0 = outrow + (t + 0) * 16;
        float* dst1 = outrow + (t + 1) * 16;
        if (!GUARD || (pbase + (t + 0) * 16 + mrow + 7 < npts)) {
            v4f d0 = {c0[0], c0[1], c0[2], c0[3]};
            v4f d1 = {c0[4], c0[5], c0[6], c0[7]};
            ((v4f*)dst0)[0] = d0;
            ((v4f*)dst0)[1] = d1;
        }
        if (!GUARD || (pbase + (t + 1) * 16 + mrow + 7 < npts)) {
            v4f d0 = {c1[0], c1[1], c1[2], c1[3]};
            v4f d1 = {c1[4], c1[5], c1[6], c1[7]};
            ((v4f*)dst1)[0] = d0;
            ((v4f*)dst1)[1] = d1;
        }
    }
}

extern "C" void kernel_launch(void* const* d_in, const int* in_sizes, int n_in,
                              void* d_out, int out_size, void* d_ws, size_t ws_size,
                              hipStream_t stream) {
    const float* x      = (const float*)d_in[0];
    const float* params = (const float*)d_in[1];
    float* out          = (float*)d_out;
    int npts = in_sizes[0];
    int blocks = (npts + NPB - 1) / NPB;
    if ((npts & (NPB - 1)) == 0) {
        kan_wmma_kernel<false><<<blocks, 256, 0, stream>>>(x, params, out, npts);
    } else {
        kan_wmma_kernel<true><<<blocks, 256, 0, stream>>>(x, params, out, npts);
    }
}
